// SimVQQuantizer_57982058496436
// MI455X (gfx1250) — compile-verified, matrix operand order checked by reading, not run on hardware
//
#include <hip/hip_runtime.h>

// Problem constants (from reference)
#define K_CODES 8192
#define DIM     256
#define NVEC    16384          // 16 * 32 * 32
#define NOUT    4194304        // 16 * 256 * 32 * 32
#define BN      128            // z rows per block in distance kernel
#define BK      128            // codebook rows per LDS chunk
#define KSPLIT  2              // K-halves (grid parallelism)
#define NCHUNKH (K_CODES / BK / KSPLIT) // 32 chunks per block

typedef __attribute__((ext_vector_type(16))) __bf16 bf16x16;
typedef __attribute__((ext_vector_type(8)))  float  f32x8;

__device__ __forceinline__ unsigned short f2bf(float f) {
  union { float f; unsigned int u; } v; v.f = f;
  unsigned int r = (v.u + 0x7fffu + ((v.u >> 16) & 1u)) >> 16;  // RNE
  return (unsigned short)r;
}

// Async global->LDS copy, 16B per lane, GVS mode (SGPR base + 32-bit VGPR
// byte offset). Tracked on ASYNCcnt, NOT DScnt -> compute-side ds waits
// never see staging traffic.
__device__ __forceinline__ void async_copy_b128(unsigned lds_off, unsigned goff,
                                                const void* sbase) {
  asm volatile("global_load_async_to_lds_b128 %0, %1, %2 offset:0"
               :: "v"(lds_off), "v"(goff), "s"(sbase) : "memory");
}
__device__ __forceinline__ void wait_async0() {
  asm volatile("s_wait_asynccnt 0x0" ::: "memory");
}

__device__ __forceinline__ float block_reduce_sum(float v) {
  #pragma unroll
  for (int m = 16; m >= 1; m >>= 1) v += __shfl_xor(v, m, 32);
  __shared__ float ps[8];
  const int lane = threadIdx.x & 31, w = threadIdx.x >> 5;
  if (lane == 0) ps[w] = v;
  __syncthreads();
  v = (threadIdx.x < 8) ? ps[threadIdx.x] : 0.0f;
  if (w == 0) {
    #pragma unroll
    for (int m = 4; m >= 1; m >>= 1) v += __shfl_xor(v, m, 32);
  }
  return v;  // valid on thread 0
}

// ---------------- init: zero pw_norm / counts / loss accumulators ----------
__global__ __launch_bounds__(256) void init_kernel(float* zero_region, int n) {
  int i = blockIdx.x * 256 + threadIdx.x;
  if (i < n) zero_region[i] = 0.0f;
}

// ---------------- proj: pw = cb @ Wᵀ + b ; bf16 copy ; ||pw||² -------------
__global__ __launch_bounds__(256) void proj_kernel(
    const float* __restrict__ cb, const float* __restrict__ pw_w,
    const float* __restrict__ pw_b, float* __restrict__ pw,
    unsigned short* __restrict__ pwb, float* __restrict__ pwn) {
  __shared__ float cbs[32 * DIM];  // 32 KB
  const int t = threadIdx.x;
  const int k0 = blockIdx.x * 32;
  #pragma unroll
  for (int i = 0; i < 32; ++i) cbs[i * DIM + t] = cb[(size_t)(k0 + i) * DIM + t];
  __syncthreads();

  float acc[32];
  const float bias = pw_b[t];
  #pragma unroll
  for (int kl = 0; kl < 32; ++kl) acc[kl] = bias;

  const float4* wrow = (const float4*)(pw_w + (size_t)t * DIM);
  for (int c4 = 0; c4 < DIM / 4; ++c4) {
    const float4 w = wrow[c4];
    #pragma unroll
    for (int kl = 0; kl < 32; ++kl) {
      const float4 c = *(const float4*)(cbs + kl * DIM + c4 * 4);
      acc[kl] += c.x * w.x + c.y * w.y + c.z * w.z + c.w * w.w;
    }
  }
  #pragma unroll
  for (int kl = 0; kl < 32; ++kl) {
    const int k = k0 + kl;
    const float v = acc[kl];
    pw[(size_t)k * DIM + t] = v;
    pwb[(size_t)k * DIM + t] = f2bf(v);
    atomicAdd(&pwn[k], v * v);
  }
}

// ---------------- zt: transpose z (B,C,H,W) -> bf16 [N, C] -----------------
__global__ __launch_bounds__(256) void zt_kernel(const float* __restrict__ z,
                                                 unsigned short* __restrict__ zb) {
  __shared__ float tile[32][33];
  const int b = blockIdx.z, ct = blockIdx.y, ht = blockIdx.x;
  const int tx = threadIdx.x & 31, ty = threadIdx.x >> 5;  // 8 rows / pass
  const float* src = z + ((size_t)b * DIM + ct * 32) * 1024 + ht * 32;
  #pragma unroll
  for (int i = 0; i < 32; i += 8) tile[ty + i][tx] = src[(size_t)(ty + i) * 1024 + tx];
  __syncthreads();
  unsigned short* dst = zb + ((size_t)(b * 1024 + ht * 32)) * DIM + ct * 32;
  #pragma unroll
  for (int i = 0; i < 32; i += 8) dst[(size_t)(ty + i) * DIM + tx] = f2bf(tile[tx][ty + i]);
}

// ---------------- distance GEMM + fused argmin (bf16 WMMA) -----------------
// dist(n,k) = ||pw_k||^2 - 2 * z_n . pw_k   (||z||^2 constant per n, dropped)
// Grid: (NVEC/BN) * KSPLIT blocks; block bx: n-tile = bx & 127, k-half = bx >> 7.
__global__ __launch_bounds__(256) void dist_argmin_kernel(
    const unsigned short* __restrict__ zb, const unsigned short* __restrict__ pwb,
    const float* __restrict__ pwn, float* __restrict__ candv,
    int* __restrict__ candi) {
  extern __shared__ char smem[];
  unsigned short* As = (unsigned short*)smem;  // BN*DIM bf16     (64 KB)
  unsigned short* Bs = As + BN * DIM;          // 2*BK*DIM bf16   (128 KB)

  const int t = threadIdx.x;
  const int wave = t >> 5, lane = t & 31;
  const int col = lane & 15, hi = lane >> 4;
  const int ntile = blockIdx.x & (NVEC / BN - 1);
  const int kh = blockIdx.x / (NVEC / BN);
  const int nblock = ntile * BN;
  const int gchunk0 = kh * NCHUNKH;  // first global chunk of this half

  // LDS byte offsets (low 32 bits of flat addr == LDS offset, ISA 10.2)
  const unsigned as_off = (unsigned)(uintptr_t)As;
  const unsigned bs_off = (unsigned)(uintptr_t)Bs;

  {  // async-stage A tile (z rows) and first B chunk of this k-half
    const unsigned goffA = (unsigned)(nblock * DIM * 2);
    const unsigned goffB = (unsigned)(gchunk0 * BK * DIM * 2);
    #pragma unroll
    for (int i = 0; i < 16; ++i) {
      const unsigned e = (unsigned)(t + i * 256) * 16u;
      async_copy_b128(as_off + e, goffA + e, zb);
      async_copy_b128(bs_off + e, goffB + e, pwb);
    }
    wait_async0();
  }
  __syncthreads();

  // A fragments pinned in registers for the whole K loop.
  // ISA 16-bit A layout: lane m holds K {0..7,16..23}; lane m+16 holds {8..15,24..31}.
  bf16x16 afrag[8];
  {
    const unsigned short* abase = As + (size_t)(wave * 16 + col) * DIM;
    #pragma unroll
    for (int cc = 0; cc < 8; ++cc) {
      union { uint4 q[2]; bf16x16 v; } u;
      const int c0 = cc * 32 + hi * 8;
      u.q[0] = *(const uint4*)(abase + c0);
      u.q[1] = *(const uint4*)(abase + c0 + 16);
      afrag[cc] = u.v;
    }
  }

  float bestv[8]; int besti[8];
  #pragma unroll
  for (int s = 0; s < 8; ++s) { bestv[s] = 3.4e38f; besti[s] = 0; }

  // deferred epilogue state: fold previous sub-pair while loads are in flight
  f32x8 pacc0 = {}, pacc1 = {};
  int pk = -1;

  int p = 0;
  for (int kc = 0; kc < NCHUNKH; ++kc) {
    const int g = gchunk0 + kc;  // global chunk id
    if (kc + 1 < NCHUNKH) {      // async-stage next chunk into the other buffer
      const unsigned gbase = (unsigned)((g + 1) * BK * DIM * 2);
      const unsigned lbase = bs_off + (unsigned)((p ^ 1) * BK * DIM * 2);
      #pragma unroll
      for (int i = 0; i < 16; ++i) {
        const unsigned e = (unsigned)(t + i * 256) * 16u;
        async_copy_b128(lbase + e, gbase + e, pwb);
      }
    }
    if (kc + 2 < NCHUNKH)  // L2 prefetch hint (global_prefetch_b8)
      __builtin_prefetch(pwb + (size_t)(g + 2) * BK * DIM + (size_t)t * 128, 0, 0);

    const unsigned short* B = Bs + (size_t)p * BK * DIM;
    const int kbase = g * BK;
    #pragma unroll
    for (int sub = 0; sub < 8; sub += 2) {
      const unsigned short* b0 = B + (size_t)(sub * 16 + col) * DIM;
      const unsigned short* b1 = b0 + 16 * DIM;
      // Issue all 32 fragment loads for this sub-pair first ...
      union BU { uint4 q[2]; bf16x16 v; } u0[8], u1[8];
      #pragma unroll
      for (int cc = 0; cc < 8; ++cc) {
        const int c0 = cc * 32 + hi * 8;
        u0[cc].q[0] = *(const uint4*)(b0 + c0);
        u0[cc].q[1] = *(const uint4*)(b0 + c0 + 16);
        u1[cc].q[0] = *(const uint4*)(b1 + c0);
        u1[cc].q[1] = *(const uint4*)(b1 + c0 + 16);
      }
      // ... fold the PREVIOUS sub-pair's argmin while they are in flight
      // (pure VALU + pwn loads; touches no fragment registers).
      if (pk >= 0) {
        const float nrm0 = pwn[pk];
        const float nrm1 = pwn[pk + 16];
        #pragma unroll
        for (int s = 0; s < 8; ++s) {
          const float d0 = nrm0 - 2.0f * pacc0[s];
          if (d0 < bestv[s]) { bestv[s] = d0; besti[s] = pk; }     // k ascending
          const float d1 = nrm1 - 2.0f * pacc1[s];
          if (d1 < bestv[s]) { bestv[s] = d1; besti[s] = pk + 16; }
        }
      }
      f32x8 acc0 = {}, acc1 = {};
      #pragma unroll
      for (int cc = 0; cc < 8; ++cc) {
        acc0 = __builtin_amdgcn_wmma_f32_16x16x32_bf16(
            false, afrag[cc], false, u0[cc].v, (short)0, acc0, false, false);
        acc1 = __builtin_amdgcn_wmma_f32_16x16x32_bf16(
            false, afrag[cc], false, u1[cc].v, (short)0, acc1, false, false);
      }
      pacc0 = acc0; pacc1 = acc1;
      pk = kbase + sub * 16 + col;  // C/D layout: slot s -> M = s + 8*hi, N = col
    }
    wait_async0();   // staging for next chunk complete (this wave)
    __syncthreads(); // ... and for all waves
    p ^= 1;
  }
  // fold the final pending sub-pair
  {
    const float nrm0 = pwn[pk];
    const float nrm1 = pwn[pk + 16];
    #pragma unroll
    for (int s = 0; s < 8; ++s) {
      const float d0 = nrm0 - 2.0f * pacc0[s];
      if (d0 < bestv[s]) { bestv[s] = d0; besti[s] = pk; }
      const float d1 = nrm1 - 2.0f * pacc1[s];
      if (d1 < bestv[s]) { bestv[s] = d1; besti[s] = pk + 16; }
    }
  }

  // reduce argmin across the 16 k-columns (lanes) of each half-wave
  #pragma unroll
  for (int m = 1; m < 16; m <<= 1) {
    #pragma unroll
    for (int s = 0; s < 8; ++s) {
      const float ov = __shfl_xor(bestv[s], m, 32);
      const int oi = __shfl_xor(besti[s], m, 32);
      if (ov < bestv[s] || (ov == bestv[s] && oi < besti[s])) { bestv[s] = ov; besti[s] = oi; }
    }
  }
  if (col == 0) {
    const int base = kh * NVEC + nblock + wave * 16 + hi * 8;
    #pragma unroll
    for (int s = 0; s < 8; ++s) { candv[base + s] = bestv[s]; candi[base + s] = besti[s]; }
  }
}

// ---------------- merge the KSPLIT per-half candidates ---------------------
__global__ __launch_bounds__(256) void merge_kernel(const float* __restrict__ cv,
                                                    const int* __restrict__ ci,
                                                    int* __restrict__ idx) {
  const int n = blockIdx.x * 256 + threadIdx.x;  // < NVEC
  float v0 = cv[n];        int i0 = ci[n];
  const float v1 = cv[NVEC + n]; const int i1 = ci[NVEC + n];
  if (v1 < v0 || (v1 == v0 && i1 < i0)) { v0 = v1; i0 = i1; }
  idx[n] = i0;
}

// ---------------- gather pw[idx] -> (B,C,H,W) output + fused MSE -----------
__global__ __launch_bounds__(256) void gather_kernel(
    const float* __restrict__ z, const float* __restrict__ pw,
    const int* __restrict__ idx, float* __restrict__ out,
    float* __restrict__ loss_acc) {
  const size_t o = (size_t)blockIdx.x * 256 + threadIdx.x;  // < NOUT
  const int hw = (int)(o & 1023);
  const int c = (int)((o >> 10) & 255);
  const int b = (int)(o >> 18);
  const int k = idx[b * 1024 + hw];
  const float q = pw[(size_t)k * DIM + c];
  const float d = z[o] - q;
  out[o] = q;
  const float s = block_reduce_sum(d * d);
  if (threadIdx.x == 0) atomicAdd(loss_acc, s);
}

// ---------------- indices-as-float + code-usage histogram ------------------
__global__ __launch_bounds__(256) void hist_kernel(const int* __restrict__ idx,
                                                   float* __restrict__ out_idx,
                                                   float* __restrict__ counts) {
  const int i = blockIdx.x * 256 + threadIdx.x;  // < NVEC
  const int k = idx[i];
  out_idx[i] = (float)k;
  atomicAdd(&counts[k], 1.0f);
}

// ---------------- finalize: perplexity + losses ----------------------------
__global__ __launch_bounds__(256) void fin_kernel(const float* __restrict__ counts,
                                                  const float* __restrict__ loss_acc,
                                                  float* __restrict__ out_scalars) {
  float s = 0.0f;
  for (int i = 0; i < K_CODES / 256; ++i) {
    const float p = counts[threadIdx.x + i * 256] * (1.0f / (float)NVEC);
    s += p * __logf(fmaxf(p, 1e-10f));
  }
  s = block_reduce_sum(s);
  if (threadIdx.x == 0) {
    const float loss = loss_acc[0] * (1.0f / (float)NOUT);
    out_scalars[0] = __expf(-s);  // perplexity
    out_scalars[1] = loss;        // loss_vq (forward value == loss_commit)
    out_scalars[2] = loss;        // loss_commit
  }
}

extern "C" void kernel_launch(void* const* d_in, const int* in_sizes, int n_in,
                              void* d_out, int out_size, void* d_ws, size_t ws_size,
                              hipStream_t stream) {
  const float* z    = (const float*)d_in[0];   // [16,256,32,32]
  const float* cbw  = (const float*)d_in[1];   // [8192,256]
  const float* pw_w = (const float*)d_in[2];   // [256,256]
  const float* pw_b = (const float*)d_in[3];   // [256]
  float* out = (float*)d_out;                  // [NOUT + NVEC + 3]
  char* ws = (char*)d_ws;

  // workspace layout (21,364,800 bytes total)
  float*          pw     = (float*)ws;                               // 8,388,608 B
  float*          pwn    = (float*)(ws + 8388608);                   //    32,768 B
  float*          counts = (float*)(ws + 8421376);                   //    32,768 B
  float*          lacc   = (float*)(ws + 8454144);                   //        64 B
  int*            idx    = (int*)(ws + 8454208);                     //    65,536 B
  unsigned short* pwb    = (unsigned short*)(ws + 8519744);          // 4,194,304 B
  unsigned short* zb     = (unsigned short*)(ws + 12714048);         // 8,388,608 B
  float*          candv  = (float*)(ws + 21102656);                  //   131,072 B
  int*            candi  = (int*)(ws + 21233728);                    //   131,072 B

  // 1) zero pwn + counts + lacc (contiguous: 16400 floats at +8388608)
  init_kernel<<<65, 256, 0, stream>>>((float*)(ws + 8388608), 16400);
  // 2) projected codebook (fp32 + bf16 + norms)
  proj_kernel<<<K_CODES / 32, 256, 0, stream>>>(cbw, pw_w, pw_b, pw, pwb, pwn);
  // 3) transpose z -> bf16 [N, C]
  zt_kernel<<<dim3(32, 8, 16), 256, 0, stream>>>(z, zb);
  // 4) fused distance GEMM + argmin (WMMA bf16 + async LDS staging), 192 KB LDS
  //    K split 2 ways -> 256 blocks for full-chip occupancy
  dist_argmin_kernel<<<(NVEC / BN) * KSPLIT, 256, 196608, stream>>>(zb, pwb, pwn,
                                                                    candv, candi);
  // 5) merge per-half argmin candidates (tie -> lower k = first occurrence)
  merge_kernel<<<NVEC / 256, 256, 0, stream>>>(candv, candi, idx);
  // 6) gather quantized output + MSE accumulation
  gather_kernel<<<NOUT / 256, 256, 0, stream>>>(z, pw, idx, out, lacc);
  // 7) indices (as float) + histogram
  hist_kernel<<<NVEC / 256, 256, 0, stream>>>(idx, out + NOUT, counts);
  // 8) perplexity + losses
  fin_kernel<<<1, 256, 0, stream>>>(counts, lacc, out + NOUT + NVEC);
}
